// EncoderLSTM_1408749273720
// MI455X (gfx1250) — matrix operand order repaired and checked by reading, not compiled, not run on hardware
//
#include <hip/hip_runtime.h>
#include <stdint.h>

// ---------------------------------------------------------------------------
// EncoderLSTM for MI455X (gfx1250): bf16 WMMA scan with LDS-resident Wh,
// VGPR-resident Wx fragments, async-DMA (ASYNCcnt) staging of per-step A
// fragments into LDS, and a software-pipelined K loop (loads run one WMMA
// quad ahead).
//   B=64, T=512, E=256, U=512.  4 batch-groups of 16 rows; each group = 8
//   workgroups x 4 waves; each wave owns 16 u-columns, all 4 gates.
//   Per step per wave: 24 K-tiles x 4 gates = 96 v_wmma_f32_16x16x32_bf16.
// ---------------------------------------------------------------------------

typedef __attribute__((ext_vector_type(16))) __bf16 v16bf;
typedef __attribute__((ext_vector_type(8)))  float  v8f;

#define B_  64
#define T_  512
#define E_  256
#define U_  512
#define NKT 24            // K tiles of 32 over K = E+U = 768
#define NKX 8             // x part (K 0..255)
#define NKH 16            // h part (K 256..767)
#define NTILES 128        // N tiles of 16 over 4U = 2048
#define NGROUP 4          // batch groups of 16 rows
#define WGPG 8            // workgroups per group
#define FRAG_ELEMS 512    // one 32x16 (or 16x32) bf16 fragment = 32 lanes x 16

// workspace layout (bytes)
#define HBUF_OFF   256
#define HBUF_BYTES (2*NGROUP*NKH*FRAG_ELEMS*2)     // 128 KB (double buffered)
#define WPACK_OFF  (HBUF_OFF + HBUF_BYTES)         // packed [Wx;Wh] bf16, 3 MB
#define WPACK_ELEMS (NTILES*NKT*FRAG_ELEMS)        // 1,572,864
#define XPACK_OFF  (WPACK_OFF + WPACK_ELEMS*2)     // packed x A-fragments, 16 MB
#define XPACK_ELEMS (T_*NGROUP*NKX*FRAG_ELEMS)     // 8,388,608

// LDS layout (dynamic shared segment starts at LDS offset 0)
#define LDS_W_BYTES   262144                       // 4 waves x 64KB Wh fragments
#define LDS_ASTAGE    LDS_W_BYTES                  // 24KB per-step A fragments
#define LDS_ASTAGE_H  (LDS_ASTAGE + 8192)          // h part of the A stage
#define LDS_TSTAGE    (LDS_ASTAGE + 24576)         // 2KB h transpose stage
#define LDS_TOTAL     (LDS_TSTAGE + 2048)          // 288,768 bytes (< 320KB WGP)

// ---------------------------------------------------------------------------

__global__ void k_zero(uint32_t* p, int n) {
  int i = blockIdx.x * blockDim.x + threadIdx.x;
  if (i < n) p[i] = 0u;
}

// Pack W = [Wx ; Wh] (768 x 2048, gates concatenated [f|i|o|g]) into B-matrix
// fragment layout: frag(nt,kt): lane l -> N = nt*16 + (l&15),
//   j<8: K = kt*32 + (l<16?0:8)+j ;  j>=8: K = kt*32 + (l<16?16:24)+(j-8)
__global__ void k_packW(const float* __restrict__ Wfx, const float* __restrict__ Wfh,
                        const float* __restrict__ Wix, const float* __restrict__ Wih,
                        const float* __restrict__ Wox, const float* __restrict__ Woh,
                        const float* __restrict__ Wgx, const float* __restrict__ Wgh,
                        uint16_t* __restrict__ wp_raw) {
  int p = blockIdx.x * blockDim.x + threadIdx.x;
  if (p >= WPACK_ELEMS) return;
  __bf16* wp = (__bf16*)wp_raw;
  int j    = p & 15;
  int lane = (p >> 4) & 31;
  int kt   = (p >> 9) % NKT;
  int nt   = (p >> 9) / NKT;
  int N = nt * 16 + (lane & 15);
  int q = N >> 9;          // gate: 0=f 1=i 2=o 3=g
  int u = N & (U_ - 1);
  int K = kt * 32 + ((j < 8) ? (((lane & 16) ? 8 : 0) + j) : (((lane & 16) ? 24 : 16) + (j - 8)));
  const float* Wx = (q == 0) ? Wfx : (q == 1) ? Wix : (q == 2) ? Wox : Wgx;
  const float* Wh = (q == 0) ? Wfh : (q == 1) ? Wih : (q == 2) ? Woh : Wgh;
  float v = (K < E_) ? Wx[(size_t)K * U_ + u] : Wh[(size_t)(K - E_) * U_ + u];
  wp[p] = (__bf16)v;
}

// Pack inputs (B,T,E) into A-matrix fragment layout per (t, group, ktile):
// lane l -> M = l&15;  j<8: K = (l<16?0:8)+j ;  j>=8: K = (l<16?16:24)+(j-8)
__global__ void k_packX(const float* __restrict__ in, uint16_t* __restrict__ xp_raw) {
  int p = blockIdx.x * blockDim.x + threadIdx.x;
  if (p >= XPACK_ELEMS) return;
  __bf16* xp = (__bf16*)xp_raw;
  int j    = p & 15;
  int lane = (p >> 4) & 31;
  int kt   = (p >> 9) & 7;
  int g    = (p >> 12) & 3;
  int t    = p >> 14;
  int m = lane & 15;
  int b = g * 16 + m;
  int koff = (j < 8) ? (((lane & 16) ? 8 : 0) + j)
                     : (((lane & 16) ? 24 : 16) + (j - 8));
  int e = kt * 32 + koff;
  xp[p] = (__bf16) in[((size_t)b * T_ + t) * E_ + e];
}

// ---------------------------------------------------------------------------

__device__ __forceinline__ float sigmoidf_(float x) {
  return 1.0f / (1.0f + __expf(-x));
}

// 16B global -> LDS async copy (VFLAT async encoding, tracked by ASYNCcnt).
__device__ __forceinline__ void async_copy16(unsigned lds_off, const void* gsrc) {
  unsigned long long ga = (unsigned long long)(uintptr_t)gsrc;
  asm volatile("global_load_async_to_lds_b128 %0, %1, off"
               :: "v"(lds_off), "v"(ga) : "memory");
}
__device__ __forceinline__ void wait_async0() {
  asm volatile("s_wait_asynccnt 0x0" ::: "memory");
}

__global__ void __launch_bounds__(128, 1)
k_lstm(const float* __restrict__ bfp, const float* __restrict__ bip,
       const float* __restrict__ bop, const float* __restrict__ bgp,
       const uint16_t* __restrict__ wp_raw, const uint16_t* __restrict__ xp_raw,
       uint16_t* __restrict__ hb_raw, unsigned* __restrict__ cnt,
       float* __restrict__ out) {
  extern __shared__ __align__(32) unsigned char smem[];
  const __bf16* wp = (const __bf16*)wp_raw;
  const __bf16* xp = (const __bf16*)xp_raw;
  __bf16* hb = (__bf16*)hb_raw;

  const int tid    = threadIdx.x;
  const int lane   = tid & 31;
  const int wave   = tid >> 5;
  const int g      = blockIdx.x / WGPG;
  const int wg     = blockIdx.x % WGPG;
  const int ucol   = lane & 15;
  const int hihalf = (lane >> 4) & 1;
  const int ub     = wg * 64 + wave * 16;   // wave's u base (16 columns)
  const int u      = ub + ucol;
  const int ntb    = wg * 4 + wave;         // u-tile index 0..31

  // Per-gate packed weight bases (gate q columns live at N-tile q*32 + ntb)
  const __bf16* wB0 = wp + (size_t)(0 * 32 + ntb) * NKT * FRAG_ELEMS;
  const __bf16* wB1 = wp + (size_t)(1 * 32 + ntb) * NKT * FRAG_ELEMS;
  const __bf16* wB2 = wp + (size_t)(2 * 32 + ntb) * NKT * FRAG_ELEMS;
  const __bf16* wB3 = wp + (size_t)(3 * 32 + ntb) * NKT * FRAG_ELEMS;

  // Preload this wave's Wh B-fragments into LDS: 4 gates x 16 ktiles x 1KB = 64KB
  {
    const __bf16* srcs[4] = {wB0, wB1, wB2, wB3};
#pragma unroll
    for (int q = 0; q < 4; ++q) {
#pragma unroll 4
      for (int kh = 0; kh < NKH; ++kh) {
        const v16bf* s = (const v16bf*)(srcs[q] + (size_t)((NKX + kh) * 32 + lane) * 16);
        v16bf* d = (v16bf*)(smem + ((wave * 64 + q * 16 + kh) * 1024) + lane * 32);
        *d = *s;
      }
    }
  }

  // Pin the x-part (K 0..255) B fragments in VGPRs for the whole scan:
  // 4 gates x 8 K-tiles x 8 VGPRs = 256 VGPRs (occupancy-1 wave, 1024 avail).
  v16bf bx0[NKX], bx1[NKX], bx2[NKX], bx3[NKX];
#pragma unroll
  for (int kt = 0; kt < NKX; ++kt) {
    const size_t fo = (size_t)(kt * 32 + lane) * 16;
    bx0[kt] = *(const v16bf*)(wB0 + fo);
    bx1[kt] = *(const v16bf*)(wB1 + fo);
    bx2[kt] = *(const v16bf*)(wB2 + fo);
    bx3[kt] = *(const v16bf*)(wB3 + fo);
  }

  const float bF = bfp[u], bI = bip[u], bO = bop[u], bG = bgp[u];
  __bf16* stage = (__bf16*)(smem + LDS_TSTAGE);   // 16 x 64 bf16 transpose stage
  v8f c = {0.f, 0.f, 0.f, 0.f, 0.f, 0.f, 0.f, 0.f};
  unsigned* mycnt = cnt + g;
  __syncthreads();

  for (int t = 0; t < T_; ++t) {
    const __bf16* xb    = xp + (size_t)((t * NGROUP + g) * NKX) * FRAG_ELEMS;
    const __bf16* hbase = hb + (size_t)(((t & 1) * NGROUP + g) * NKH) * FRAG_ELEMS;

    // ---- Stage this step's A fragments into LDS with async DMA -----------
    // x slab: 8KB, h slab: 16KB; 128 threads x 16B per issue.
    {
      const char* xsrc = (const char*)xb;
#pragma unroll
      for (int i = 0; i < 4; ++i) {
        const unsigned off = (unsigned)(tid * 16 + i * 2048);
        async_copy16(LDS_ASTAGE + off, xsrc + off);
      }
      const char* hsrc = (const char*)hbase;
#pragma unroll
      for (int i = 0; i < 8; ++i) {
        const unsigned off = (unsigned)(tid * 16 + i * 2048);
        async_copy16(LDS_ASTAGE_H + off, hsrc + off);
      }
      wait_async0();
      __syncthreads();
    }

    v8f aF = {bF, bF, bF, bF, bF, bF, bF, bF};
    v8f aI = {bI, bI, bI, bI, bI, bI, bI, bI};
    v8f aO = {bO, bO, bO, bO, bO, bO, bO, bO};
    v8f aG = {bG, bG, bG, bG, bG, bG, bG, bG};

    // Unified K loop over 24 tiles (A stage is contiguous: x slab then h slab)
    // Software-pipelined: loads for tile k+1 issue while the quad for k runs,
    // so the s_wait before each quad covers loads issued one iteration back.
    v16bf a_cur = *(const v16bf*)(smem + LDS_ASTAGE + lane * 32);
    v16bf b0c = bx0[0], b1c = bx1[0], b2c = bx2[0], b3c = bx3[0];
#pragma unroll
    for (int k = 0; k < NKT; ++k) {
      v16bf a_n = a_cur, b0n = b0c, b1n = b1c, b2n = b2c, b3n = b3c;
      if (k + 1 < NKT) {
        a_n = *(const v16bf*)(smem + LDS_ASTAGE + (k + 1) * 1024 + lane * 32);
        if (k + 1 < NKX) {         // compile-time resolved under full unroll
          b0n = bx0[k + 1]; b1n = bx1[k + 1]; b2n = bx2[k + 1]; b3n = bx3[k + 1];
        } else {
          const unsigned lb = (unsigned)((wave * 64 + (k + 1 - NKX)) * 1024 + lane * 32);
          b0n = *(const v16bf*)(smem + lb + 0u * 16384u);
          b1n = *(const v16bf*)(smem + lb + 1u * 16384u);
          b2n = *(const v16bf*)(smem + lb + 2u * 16384u);
          b3n = *(const v16bf*)(smem + lb + 3u * 16384u);
        }
      }
      aF = __builtin_amdgcn_wmma_f32_16x16x32_bf16(false, a_cur, false, b0c, (short)0, aF, false, false);
      aI = __builtin_amdgcn_wmma_f32_16x16x32_bf16(false, a_cur, false, b1c, (short)0, aI, false, false);
      aO = __builtin_amdgcn_wmma_f32_16x16x32_bf16(false, a_cur, false, b2c, (short)0, aO, false, false);
      aG = __builtin_amdgcn_wmma_f32_16x16x32_bf16(false, a_cur, false, b3c, (short)0, aG, false, false);
      a_cur = a_n; b0c = b0n; b1c = b1n; b2c = b2n; b3c = b3n;
    }

    // Prefetch next step's x A-fragments toward L2 (global_prefetch_b8).
    if (t + 1 < T_) {
      const __bf16* xn = xp + (size_t)(((t + 1) * NGROUP + g) * NKX) * FRAG_ELEMS;
#pragma unroll
      for (int kt = 0; kt < NKX; ++kt)
        __builtin_prefetch((const void*)(xn + (size_t)(kt * 32 + lane) * 16), 0, 1);
    }

    // Elementwise gate math; c stays in VGPRs.
    float hv[8];
#pragma unroll
    for (int r = 0; r < 8; ++r) {
      float fv = sigmoidf_(aF[r]);
      float iv = sigmoidf_(aI[r]);
      float ov = sigmoidf_(aO[r]);
      float gv = sigmoidf_(aG[r]);
      float cv = fv * c[r] + iv * gv;
      c[r] = cv;
      hv[r] = tanhf(cv) * ov;
    }

    // Emit output (f32) and stage h (bf16) for transpose.
#pragma unroll
    for (int r = 0; r < 8; ++r) {
      const int brow = g * 16 + r + hihalf * 8;           // C/D layout: M = r (+8)
      out[((size_t)brow * T_ + t) * U_ + u] = hv[r];
      stage[(r + hihalf * 8) * 64 + (wave * 16 + ucol)] = (__bf16)hv[r];
    }
    __syncthreads();

    // Repack h(t+1) tile (16 x 64) into 2 A-fragments in the global h buffer.
    {
      const int fi  = tid >> 6;          // which of the WG's 2 K-tiles
      const int rem = tid & 63;
      const int l   = rem >> 1;          // fragment lane 0..31
      const int j0  = (rem & 1) * 8;     // low/high 8 halves of that lane
      __align__(16) __bf16 tmp[8];
#pragma unroll
      for (int jj = 0; jj < 8; ++jj) {
        int j = j0 + jj;
        int m = l & 15;
        int koff = (j < 8) ? (((l & 16) ? 8 : 0) + j)
                           : (((l & 16) ? 24 : 16) + (j - 8));
        tmp[jj] = stage[m * 64 + fi * 32 + koff];
      }
      __bf16* dst = hb + (size_t)((((t + 1) & 1) * NGROUP + g) * NKH + (wg * 2 + fi)) * FRAG_ELEMS
                       + l * 16 + j0;
      *(uint4*)dst = *(const uint4*)tmp;
    }

    // Inter-workgroup step barrier (monotonic counter, release/acquire).
    __threadfence();
    __syncthreads();
    if (tid == 0)
      __hip_atomic_fetch_add(mycnt, 1u, __ATOMIC_RELEASE, __HIP_MEMORY_SCOPE_AGENT);
    const unsigned tgt = (unsigned)(WGPG * (t + 1));
    while (__hip_atomic_load(mycnt, __ATOMIC_ACQUIRE, __HIP_MEMORY_SCOPE_AGENT) < tgt) { }
  }
}

// ---------------------------------------------------------------------------

extern "C" void kernel_launch(void* const* d_in, const int* in_sizes, int n_in,
                              void* d_out, int out_size, void* d_ws, size_t ws_size,
                              hipStream_t stream) {
  const float* inputs = (const float*)d_in[0];
  const float* Wfx = (const float*)d_in[1];
  const float* Wfh = (const float*)d_in[2];
  const float* bfp = (const float*)d_in[3];
  const float* Wix = (const float*)d_in[4];
  const float* Wih = (const float*)d_in[5];
  const float* bip = (const float*)d_in[6];
  const float* Wox = (const float*)d_in[7];
  const float* Woh = (const float*)d_in[8];
  const float* bop = (const float*)d_in[9];
  const float* Wgx = (const float*)d_in[10];
  const float* Wgh = (const float*)d_in[11];
  const float* bgp = (const float*)d_in[12];

  uint8_t* ws = (uint8_t*)d_ws;
  unsigned*  cnt = (unsigned*)ws;                       // 4 x u32 step counters
  uint16_t*  hbf = (uint16_t*)(ws + HBUF_OFF);          // 128 KB h fragments
  uint16_t*  wpk = (uint16_t*)(ws + WPACK_OFF);         // 3 MB packed W
  uint16_t*  xpk = (uint16_t*)(ws + XPACK_OFF);         // 16 MB packed x
  float* out = (float*)d_out;

  // Zero counters + h double buffer every call (deterministic, ws is poisoned).
  const int nz = (HBUF_OFF + HBUF_BYTES) / 4;
  k_zero<<<(nz + 255) / 256, 256, 0, stream>>>((uint32_t*)ws, nz);

  k_packW<<<(WPACK_ELEMS + 255) / 256, 256, 0, stream>>>(Wfx, Wfh, Wix, Wih,
                                                         Wox, Woh, Wgx, Wgh, wpk);
  k_packX<<<(XPACK_ELEMS + 255) / 256, 256, 0, stream>>>(inputs, xpk);

  // 256 KB LDS weight cache + 24 KB A stage + 2 KB transpose stage.
  k_lstm<<<NGROUP * WGPG, 128, LDS_TOTAL, stream>>>(bfp, bip, bop, bgp,
                                                    wpk, xpk, hbf, cnt, out);
}
